// GatherVertical_40656160424516
// MI455X (gfx1250) — matrix operand order repaired and checked by reading, not compiled, not run on hardware
//
#include <hip/hip_runtime.h>

// Problem constants (from reference): B=4096, CIN=128, COUT=128, 9 offsets x 9 cols.
#define NB   4096
#define CIN  128
#define COUT 128
#define NK   9
#define NX   9

typedef __attribute__((ext_vector_type(16))) __bf16          v16bf;
typedef __attribute__((ext_vector_type(8)))  float           v8f;
typedef __attribute__((ext_vector_type(16))) unsigned short  v16u;
typedef __attribute__((ext_vector_type(8)))  unsigned short  u8s;

__device__ __forceinline__ unsigned short f2bf_rne(float f) {
  unsigned int u = __float_as_uint(f);
  u += 0x7fffu + ((u >> 16) & 1u);          // round-to-nearest-even into bf16
  return (unsigned short)(u >> 16);
}
__device__ __forceinline__ float bf2f(unsigned short h) {
  return __uint_as_float(((unsigned int)h) << 16);
}

// One workgroup = 8 waves = 128 oc x 128 m tile of one k-offset GEMM.
// Each wave: 8 accumulator tiles (16 oc x 16 m each), K-loop = 4 chunks of 32.
// fp32 accuracy via bf16 hi/lo split: acc += AH*BH + AH*BL + AL*BH.
__global__ __launch_bounds__(256) void gather_vertical_wmma(
    const float* __restrict__ in,    // [B, CIN, 9, 9]
    const float* __restrict__ w,     // [9, COUT, CIN]
    const float* __restrict__ bias,  // [9, COUT]
    float* __restrict__ out)         // [B, 9, COUT, 9]
{
  __shared__ __attribute__((aligned(16))) unsigned short sWhi[COUT * CIN];
  __shared__ __attribute__((aligned(16))) unsigned short sWlo[COUT * CIN];

  const int k   = blockIdx.y;
  const int tid = threadIdx.x;

  // ---- stage W[k] (A matrix, row-major [oc][i]) into LDS as bf16 hi/lo ----
  const float* wk = w + k * (COUT * CIN);
#pragma unroll
  for (int j = 0; j < (COUT * CIN) / 256; ++j) {
    const int idx = tid + j * 256;            // idx = oc*128 + i (coalesced)
    const float f = wk[idx];
    const unsigned short h = f2bf_rne(f);
    sWhi[idx] = h;
    sWlo[idx] = f2bf_rne(f - bf2f(h));
  }
  __syncthreads();

  const int wave = tid >> 5;
  const int lane = tid & 31;
  const int half = lane >> 4;   // which 16-lane half of the wave
  const int l16  = lane & 15;

  // N (column) index: m = b*9 + x
  const int m = blockIdx.x * 128 + wave * 16 + l16;
  const int b = m / NX;
  const int x = m - b * NX;
  const int inBase  = b * (CIN * NK * NX) + k * NX + x;           // + i*81
  const int outBase = b * (NK * COUT * NX) + k * (COUT * NX) + x; // + oc*9

  // ---- B fragments (inputs), per K-chunk of 32, bf16 hi/lo split.
  // Dense B 32x16 layout: lanes 0-15 hold K={0..7,16..23}, lanes 16-31 hold K={8..15,24..31}.
  v16u bhU[4], blU[4];
  const int bK0 = half * 8;
#pragma unroll
  for (int kc = 0; kc < 4; ++kc) {
    v16u bh, bl;
#pragma unroll
    for (int e = 0; e < 16; ++e) {
      const int i = kc * 32 + bK0 + (e < 8 ? e : e + 8);
      const float f = in[inBase + i * (NK * NX)];
      const unsigned short h = f2bf_rne(f);
      bh[e] = h;
      bl[e] = f2bf_rne(f - bf2f(h));
    }
    bhU[kc] = bh;
    blU[kc] = bl;
  }

  v8f acc[8] = {};

  // ---- main compute: 8 oc-tiles x 4 K-chunks x 3 split terms = 96 WMMAs ----
  // A 16-bit 16x32 layout: lanes 0-15 K={0..7,16..23}, lanes 16-31 K={8..15,24..31}.
  const int aK0 = half * 8;
#pragma unroll
  for (int mt = 0; mt < 8; ++mt) {
    const int oc = mt * 16 + l16;
#pragma unroll
    for (int kc = 0; kc < 4; ++kc) {
      const int off0 = oc * CIN + kc * 32 + aK0;  // elements e=0..7
      const int off1 = off0 + 16;                 // elements e=8..15
      const u8s ah0 = *(const u8s*)&sWhi[off0];
      const u8s ah1 = *(const u8s*)&sWhi[off1];
      const u8s al0 = *(const u8s*)&sWlo[off0];
      const u8s al1 = *(const u8s*)&sWlo[off1];
      const v16u ahU = __builtin_shufflevector(ah0, ah1,
          0, 1, 2, 3, 4, 5, 6, 7, 8, 9, 10, 11, 12, 13, 14, 15);
      const v16u alU = __builtin_shufflevector(al0, al1,
          0, 1, 2, 3, 4, 5, 6, 7, 8, 9, 10, 11, 12, 13, 14, 15);
      const v16bf AH = __builtin_bit_cast(v16bf, ahU);
      const v16bf AL = __builtin_bit_cast(v16bf, alU);
      const v16bf BH = __builtin_bit_cast(v16bf, bhU[kc]);
      const v16bf BL = __builtin_bit_cast(v16bf, blU[kc]);
      acc[mt] = __builtin_amdgcn_wmma_f32_16x16x32_bf16(
          false, AH, false, BH, (short)0, acc[mt], false, false);
      acc[mt] = __builtin_amdgcn_wmma_f32_16x16x32_bf16(
          false, AH, false, BL, (short)0, acc[mt], false, false);
      acc[mt] = __builtin_amdgcn_wmma_f32_16x16x32_bf16(
          false, AL, false, BH, (short)0, acc[mt], false, false);
    }
  }

  // ---- bias + store. C/D 16x16 f32: VGPR r -> row r (lanes 0-15) / r+8
  // (lanes 16-31), column = l16 = m -> unit-stride stores along x.
  const float* bk = bias + k * COUT;
#pragma unroll
  for (int mt = 0; mt < 8; ++mt) {
#pragma unroll
    for (int r = 0; r < 8; ++r) {
      const int oc = mt * 16 + r + half * 8;
      out[outBase + oc * NX] = acc[mt][r] + bk[oc];
    }
  }
}

extern "C" void kernel_launch(void* const* d_in, const int* in_sizes, int n_in,
                              void* d_out, int out_size, void* d_ws, size_t ws_size,
                              hipStream_t stream) {
  (void)in_sizes; (void)n_in; (void)out_size; (void)d_ws; (void)ws_size;
  const float* in   = (const float*)d_in[0];  // [4096,128,9,9]
  const float* w    = (const float*)d_in[1];  // [9,128,128]
  const float* bias = (const float*)d_in[2];  // [9,128]
  float* out = (float*)d_out;                 // [4096,9,128,9]

  dim3 grid((NB * NX) / 128, NK);  // 288 m-blocks x 9 offsets
  gather_vertical_wmma<<<grid, 256, 0, stream>>>(in, w, bias, out);
}
